// ModularGNN_10514079941543
// MI455X (gfx1250) — compile-verified, matrix operand order checked by reading, not run on hardware
//
#include <hip/hip_runtime.h>
#include <hip/hip_bf16.h>
#include <stdint.h>

#define D_H  256
#define EPS  1e-5f

typedef _Float16 half_t;
typedef half_t v16h __attribute__((ext_vector_type(16)));
typedef float  v8f  __attribute__((ext_vector_type(8)));

__device__ __forceinline__ float gelu_f(float x) {
  float x3 = x * x * x;
  return 0.5f * x * (1.0f + tanhf(0.7978845608028654f * (x + 0.044715f * x3)));
}

// ---------------- utility kernels ----------------
__global__ void k_zero(float* p, long n) {
  long i = (long)blockIdx.x * blockDim.x + threadIdx.x;
  if (i < n) p[i] = 0.0f;
}

__global__ void k_degree(const long long* __restrict__ dst, float* __restrict__ deg, int E) {
  int e = blockIdx.x * blockDim.x + threadIdx.x;
  if (e < E) atomicAdd(&deg[(int)dst[e]], 1.0f);
}

__global__ void k_deg_fin(const float* __restrict__ deg, float* __restrict__ isd,
                          float* __restrict__ invd, int N) {
  int i = blockIdx.x * blockDim.x + threadIdx.x;
  if (i < N) {
    float d = deg[i] + 1.0f;   // +1 self loop
    isd[i]  = rsqrtf(d);
    invd[i] = 1.0f / d;
  }
}

__global__ void k_f32_to_f16(const float* __restrict__ s, half_t* __restrict__ d, long n) {
  long i = (long)blockIdx.x * blockDim.x + threadIdx.x;
  if (i < n) d[i] = (half_t)s[i];
}

// w: [K x 256] row-major (f32)  ->  wt: [256 x K] (f16), so each output column is contiguous
__global__ void k_transpose_w(const float* __restrict__ w, half_t* __restrict__ wt, int K) {
  int gid = blockIdx.x * blockDim.x + threadIdx.x;    // gid = k*256 + n
  if (gid < K * 256) {
    int n = gid & 255;
    int k = gid >> 8;
    wt[(size_t)n * K + k] = (half_t)w[gid];
  }
}

// ---------------- WMMA GEMM: C[M x 256] = A[M x K] * W[K x 256] ----------------
// A: f16 row-major. Wt: f16, [256 x K] (column n of W contiguous over K).
// grid.x = M/16 (M padded to multiple of 16). block = 256 threads = 8 waves.
// Each wave computes two 16x16 output tiles -> 8 waves * 2 = 16 tiles = 256 cols.
__global__ __launch_bounds__(256) void k_gemm_wmma(
    const half_t* __restrict__ A,
    const half_t* __restrict__ Wt,
    float* __restrict__ C,
    int K) {
  const int rowTile = blockIdx.x * 16;
  const int wave = threadIdx.x >> 5;
  const int lane = threadIdx.x & 31;
  const int m  = lane & 15;   // row (A) / col (B) within tile
  const int hi = lane >> 4;   // lane-half selects K sub-range
  const half_t* __restrict__ arow = A + (size_t)(rowTile + m) * K;
#pragma unroll
  for (int ct = 0; ct < 2; ++ct) {
    const int colTile = (wave * 2 + ct) * 16;
    const half_t* __restrict__ bcol = Wt + (size_t)(colTile + m) * K;
    v8f acc = {};
    for (int k0 = 0; k0 < K; k0 += 32) {
      v16h a, b;
#pragma unroll
      for (int v = 0; v < 8; ++v) {
        // 16-bit A/B operand layout (ISA 7.12.2): lanes 0-15 K=0..7 / 16..23,
        // lanes 16-31 K=8..15 / 24..31, two halves per VGPR.
        const int kk = k0 + ((v >> 2) << 4) + (hi << 3) + ((v & 3) << 1);
        a[2 * v]     = arow[kk];
        a[2 * v + 1] = arow[kk + 1];
        b[2 * v]     = bcol[kk];
        b[2 * v + 1] = bcol[kk + 1];
      }
      acc = __builtin_amdgcn_wmma_f32_16x16x32_f16(false, a, false, b,
                                                   (short)0, acc, false, false);
    }
    // C/D layout: lane L, VGPR v -> row = v + 8*hi, col = lane&15
#pragma unroll
    for (int v = 0; v < 8; ++v)
      C[(size_t)(rowTile + v + 8 * hi) * D_H + colTile + m] = acc[v];
  }
}

// ---------------- edge scatter: agg[dst] += hw[src] * isd[src]*isd[dst] ----------------
// one block (256 threads) per edge; thread = feature -> coalesced atomics per dst row
__global__ __launch_bounds__(256) void k_scatter(
    const long long* __restrict__ src, const long long* __restrict__ dst,
    const float* __restrict__ isd, const float* __restrict__ hw,
    float* __restrict__ agg) {
  const int e = blockIdx.x;
  const int f = threadIdx.x;
  const int s = (int)src[e];
  const int d = (int)dst[e];
  const float coef = isd[s] * isd[d];
  atomicAdd(&agg[(size_t)d * D_H + f], hw[(size_t)s * D_H + f] * coef);
}

// ---------------- combine (in place into q) + BN statistics ----------------
// q = agg + hw*invd + bias ; accumulate per-feature sum / sumsq
__global__ __launch_bounds__(256) void k_combine_stats(
    float* __restrict__ q, const float* __restrict__ p,
    const float* __restrict__ invd, const float* __restrict__ bias,
    float* __restrict__ ssum, float* __restrict__ ssq, int N) {
  const int f = threadIdx.x;                 // feature
  const int r0 = blockIdx.x * 128;
  const int rend = min(r0 + 128, N);
  const float b = bias[f];
  float ls = 0.0f, lq = 0.0f;
  for (int r = r0; r < rend; ++r) {
    const size_t idx = (size_t)r * D_H + f;
    const float v = q[idx] + p[idx] * invd[r] + b;
    q[idx] = v;
    ls += v;
    lq += v * v;
  }
  atomicAdd(&ssum[f], ls);
  atomicAdd(&ssq[f], lq);
}

// ---------------- batchnorm apply + gelu + residual ----------------
__global__ void k_bn_gelu_res(const float* __restrict__ q,
                              const float* __restrict__ ssum, const float* __restrict__ ssq,
                              const float* __restrict__ g, const float* __restrict__ bb,
                              const float* __restrict__ res, float* __restrict__ h,
                              long n, float invN) {
  long i = (long)blockIdx.x * blockDim.x + threadIdx.x;
  if (i < n) {
    const int f = (int)(i & (D_H - 1));
    const float mu  = ssum[f] * invN;
    const float var = ssq[f] * invN - mu * mu;
    const float xn  = (q[i] - mu) * rsqrtf(var + EPS) * g[f] + bb[f];
    h[i] = gelu_f(xn) + res[i];
  }
}

__global__ void k_add_bias(float* __restrict__ p, const float* __restrict__ b, long n) {
  long i = (long)blockIdx.x * blockDim.x + threadIdx.x;
  if (i < n) p[i] += b[i & (D_H - 1)];
}

__global__ void k_bias_gelu(const float* __restrict__ p, const float* __restrict__ b,
                            float* __restrict__ h, long n) {
  long i = (long)blockIdx.x * blockDim.x + threadIdx.x;
  if (i < n) h[i] = gelu_f(p[i] + b[i & (D_H - 1)]);
}

// ---------------- head: out[i] = mask[i] ? h[i,:] . w + b : 0 ----------------
__global__ void k_head(const float* __restrict__ h, const float* __restrict__ w,
                       const float* __restrict__ hb, const unsigned char* __restrict__ mask,
                       float* __restrict__ out, int N) {
  int i = blockIdx.x * blockDim.x + threadIdx.x;
  if (i < N) {
    const float4* hp = (const float4*)(h + (size_t)i * D_H);
    const float4* wp = (const float4*)w;
    float s = 0.0f;
#pragma unroll 8
    for (int j = 0; j < D_H / 4; ++j) {
      float4 a = hp[j], c = wp[j];
      s += a.x * c.x + a.y * c.y + a.z * c.z + a.w * c.w;
    }
    out[i] = mask[i] ? (s + hb[0]) : 0.0f;
  }
}

// ---------------- launch ----------------
extern "C" void kernel_launch(void* const* d_in, const int* in_sizes, int n_in,
                              void* d_out, int out_size, void* d_ws, size_t ws_size,
                              hipStream_t stream) {
  const float*     x      = (const float*)d_in[0];
  const long long* ei     = (const long long*)d_in[1];
  const unsigned char* mask = (const unsigned char*)d_in[2];
  const float* conv_w[3]  = {(const float*)d_in[3], (const float*)d_in[5], (const float*)d_in[7]};
  const float* conv_b[3]  = {(const float*)d_in[4], (const float*)d_in[6], (const float*)d_in[8]};
  const float* bn_g[3]    = {(const float*)d_in[9],  (const float*)d_in[11], (const float*)d_in[13]};
  const float* bn_b[3]    = {(const float*)d_in[10], (const float*)d_in[12], (const float*)d_in[14]};
  const float* proj_w     = (const float*)d_in[15];
  const float* proj_b     = (const float*)d_in[16];
  const float* lin_w[2]   = {(const float*)d_in[17], (const float*)d_in[19]};
  const float* lin_b[2]   = {(const float*)d_in[18], (const float*)d_in[20]};
  const float* head_w     = (const float*)d_in[21];
  const float* head_b     = (const float*)d_in[22];
  float* out = (float*)d_out;

  const int N    = in_sizes[2];        // mask has one entry per node
  const int E    = in_sizes[1] / 2;    // edge_index is [2, E]
  const int D_IN = in_sizes[0] / N;    // 128
  const int Mp   = (N + 15) & ~15;     // pad rows to WMMA tile

  // ---- carve workspace ----
  uint8_t* wsp = (uint8_t*)d_ws;
  auto carve = [&](size_t bytes) -> void* {
    void* p = (void*)wsp;
    wsp += (bytes + 255) & ~(size_t)255;
    return p;
  };
  half_t* h16  = (half_t*)carve((size_t)Mp * D_H * sizeof(half_t));
  half_t* wt16 = (half_t*)carve((size_t)D_H * D_H * sizeof(half_t));
  float*  P    = (float*)carve((size_t)Mp * D_H * sizeof(float));   // hw
  float*  Q    = (float*)carve((size_t)Mp * D_H * sizeof(float));   // agg / pre-BN
  float*  H    = (float*)carve((size_t)Mp * D_H * sizeof(float));   // current h
  float*  R    = (float*)carve((size_t)Mp * D_H * sizeof(float));   // proj residual
  float*  deg  = (float*)carve((size_t)N * sizeof(float));
  float*  isd  = (float*)carve((size_t)N * sizeof(float));
  float*  invd = (float*)carve((size_t)N * sizeof(float));
  float*  ssum = (float*)carve(512 * sizeof(float));
  float*  ssq  = ssum + 256;

  const long nND = (long)N * D_H;
  const int TB = 256;
  auto blocks = [](long n, int tb) { return (unsigned)((n + tb - 1) / tb); };
  const float invN = 1.0f / (float)N;

  const long long* srcIdx = ei;
  const long long* dstIdx = ei + E;

  // degree + normalization coefficients
  k_zero<<<blocks(N, TB), TB, 0, stream>>>(deg, N);
  k_degree<<<blocks(E, TB), TB, 0, stream>>>(dstIdx, deg, E);
  k_deg_fin<<<blocks(N, TB), TB, 0, stream>>>(deg, isd, invd, N);

  // residual: R = x @ proj_w + proj_b   (also leaves h16 = f16(x) for layer 0)
  k_f32_to_f16<<<blocks((long)N * D_IN, TB), TB, 0, stream>>>(x, h16, (long)N * D_IN);
  k_transpose_w<<<blocks((long)D_IN * D_H, TB), TB, 0, stream>>>(proj_w, wt16, D_IN);
  k_gemm_wmma<<<Mp / 16, 256, 0, stream>>>(h16, wt16, R, D_IN);
  k_add_bias<<<blocks(nND, TB), TB, 0, stream>>>(R, proj_b, nND);

  // 3 GCN layers
  for (int layer = 0; layer < 3; ++layer) {
    const int K = (layer == 0) ? D_IN : D_H;
    if (layer > 0)
      k_f32_to_f16<<<blocks(nND, TB), TB, 0, stream>>>(H, h16, nND);
    k_transpose_w<<<blocks((long)K * D_H, TB), TB, 0, stream>>>(conv_w[layer], wt16, K);
    k_gemm_wmma<<<Mp / 16, 256, 0, stream>>>(h16, wt16, P, K);

    k_zero<<<blocks(nND, TB), TB, 0, stream>>>(Q, nND);
    k_scatter<<<E, 256, 0, stream>>>(srcIdx, dstIdx, isd, P, Q);

    k_zero<<<blocks(512, TB), TB, 0, stream>>>(ssum, 512);
    k_combine_stats<<<blocks((long)N, 128) , 256, 0, stream>>>(Q, P, invd, conv_b[layer],
                                                               ssum, ssq, N);
    const float* res = (layer == 0) ? R : H;   // H read+write same element: safe
    k_bn_gelu_res<<<blocks(nND, TB), TB, 0, stream>>>(Q, ssum, ssq, bn_g[layer], bn_b[layer],
                                                      res, H, nND, invN);
  }

  // 2 linear layers: H = gelu(H @ lin_w + lin_b)
  for (int l = 0; l < 2; ++l) {
    k_f32_to_f16<<<blocks(nND, TB), TB, 0, stream>>>(H, h16, nND);
    k_transpose_w<<<blocks((long)D_H * D_H, TB), TB, 0, stream>>>(lin_w[l], wt16, D_H);
    k_gemm_wmma<<<Mp / 16, 256, 0, stream>>>(h16, wt16, P, D_H);
    k_bias_gelu<<<blocks(nND, TB), TB, 0, stream>>>(P, lin_b[l], H, nND);
  }

  // head + mask
  k_head<<<blocks(N, TB), TB, 0, stream>>>(H, head_w, head_b, mask, out, N);
}
